// CrossAttentionBlock_31920196944497
// MI455X (gfx1250) — matrix operand order repaired and checked by reading, not compiled, not run on hardware
//
#include <hip/hip_runtime.h>
#include <hip/hip_fp16.h>
#include <math.h>

typedef __attribute__((ext_vector_type(16))) _Float16 v16h;
typedef __attribute__((ext_vector_type(8)))  _Float16 v8h;
typedef __attribute__((ext_vector_type(4)))  _Float16 v4h;
typedef __attribute__((ext_vector_type(8)))  float    v8f;

#define DIM    768
#define HEADS  12
#define DKH    64
#define SEQ    2048
#define NBATCH 2
#define MROWS  (NBATCH*SEQ)   // 4096

// ---- CDNA5 async global->LDS copy (ASYNCcnt-tracked), per-lane 16B ----------
// VDST = 32-bit LDS byte offset, VADDR = 64-bit global address, "off" = no
// SADDR. INST_OFFSET is added to BOTH the LDS and the global address (ISA
// 08_async_tensor 4.4), so consecutive 16B chunks share base registers.
#define ASYNC_B128(loff, gptr, imm)                                         \
  asm volatile("global_load_async_to_lds_b128 %0, %1, off offset:%2"        \
               :: "v"(loff), "v"(gptr), "i"(imm) : "memory")
#define WAIT_ASYNC(n) asm volatile("s_wait_asynccnt " #n ::: "memory")

__device__ __forceinline__ unsigned lds_off(const void* p) {
  return (unsigned)(unsigned long long)p;   // flat addr low 32 bits = LDS offset
}

__device__ __forceinline__ v8f vzero8() {
  v8f z;
#pragma unroll
  for (int i = 0; i < 8; ++i) z[i] = 0.0f;
  return z;
}

__device__ __forceinline__ v8f wmma16(v16h a, v16h b, v8f c) {
  // D = A(16x32 f16) * B(32x16 f16) + C(16x16 f32)
  return __builtin_amdgcn_wmma_f32_16x16x32_f16(false, a, false, b, (short)0, c,
                                                false, false);
}

// A-operand fragment: 16 rows x 32 K from LDS tile (row-major, `stride` halfs/row).
// A layout (matches frag_b): lanes<16 hold K=0..15 contiguous, lanes>=16 K=16..31.
__device__ __forceinline__ v16h frag_a(const _Float16* lds, int row_base, int stride) {
  int lane = threadIdx.x & 31;
  int row  = row_base + (lane & 15);
  int k0   = (lane < 16) ? 0 : 16;
  const _Float16* p = lds + row * stride + k0;
  v8h lo = *(const v8h*)(p);
  v8h hi = *(const v8h*)(p + 8);
  return __builtin_shufflevector(lo, hi, 0,1,2,3,4,5,6,7,8,9,10,11,12,13,14,15);
}

// B-operand fragment: 32 K x 16 cols; LDS holds B^T row-major (row = output col).
// Lane<16 holds col=lane&15, K=0..15 contiguous; lane>=16 holds K=16..31.
__device__ __forceinline__ v16h frag_b(const _Float16* lds, int col_base, int stride) {
  int lane = threadIdx.x & 31;
  int col  = col_base + (lane & 15);
  int k0   = (lane < 16) ? 0 : 16;
  return *(const v16h*)(lds + col * stride + k0);
}

// ---------------- LayerNorm (fp32 in) -> f16 normalized activations ----------
__global__ __launch_bounds__(256) void ln_f16_kernel(
    const float* __restrict__ x, const float* __restrict__ w,
    const float* __restrict__ b, _Float16* __restrict__ out)
{
  const int row = blockIdx.x;
  const float* xr = x + (size_t)row * DIM;
  const int t = threadIdx.x;
  float v0 = xr[t], v1 = xr[t + 256], v2 = xr[t + 512];
  float s = v0 + v1 + v2;
  float q = v0*v0 + v1*v1 + v2*v2;
  __shared__ float red[16];
  const int lane = t & 31, wv = t >> 5;
#pragma unroll
  for (int off = 16; off > 0; off >>= 1) {
    s += __shfl_down(s, off, 32);
    q += __shfl_down(q, off, 32);
  }
  if (lane == 0) { red[wv] = s; red[8 + wv] = q; }
  __syncthreads();
  if (wv == 0) {
    s = (lane < 8) ? red[lane] : 0.0f;
    q = (lane < 8) ? red[8 + lane] : 0.0f;
#pragma unroll
    for (int off = 4; off > 0; off >>= 1) {
      s += __shfl_down(s, off, 32);
      q += __shfl_down(q, off, 32);
    }
    if (lane == 0) { red[0] = s; red[1] = q; }
  }
  __syncthreads();
  float mu  = red[0] * (1.0f / DIM);
  float var = red[1] * (1.0f / DIM) - mu * mu;
  float rs  = rsqrtf(var + 1e-5f);
  _Float16* orow = out + (size_t)row * DIM;
  orow[t]       = (_Float16)((v0 - mu) * rs * w[t]       + b[t]);
  orow[t + 256] = (_Float16)((v1 - mu) * rs * w[t + 256] + b[t + 256]);
  orow[t + 512] = (_Float16)((v2 - mu) * rs * w[t + 512] + b[t + 512]);
}

// ---------------- fp32 -> f16 weight conversion ------------------------------
__global__ void cvt_f16x4_kernel(const float4* __restrict__ in,
                                 _Float16* __restrict__ out, int n4)
{
  int i = blockIdx.x * blockDim.x + threadIdx.x;
  if (i < n4) {
    float4 v = in[i];
    v4h o;
    o[0] = (_Float16)v.x; o[1] = (_Float16)v.y;
    o[2] = (_Float16)v.z; o[3] = (_Float16)v.w;
    *(v4h*)(out + (size_t)i * 4) = o;
  }
}

// ---------------- WMMA GEMM: C[M,768] = A[M,768] @ W[768,768]^T + bias -------
// MODE 0: f16 output scattered to [B,H,N,DK]; MODE 1: fp32 row-major output.
// Block = 128 threads (4 waves); tile 128(M) x 64(N) x 32(K); wave = 32 rows.
// Double-buffered LDS fed by async global->LDS copies (6 x b128 per thread per
// k-step: 4 for the A row, 2 for the W half-row), chunk offsets in INST_OFFSET.
template <int MODE>
__global__ __launch_bounds__(128) void gemm_wmma_kernel(
    const _Float16* __restrict__ A, const _Float16* __restrict__ W,
    const float* __restrict__ bias, _Float16* __restrict__ outh,
    float* __restrict__ outf)
{
  __shared__ alignas(32) _Float16 sA[2][128 * 32];
  __shared__ alignas(32) _Float16 sB[2][64 * 32];
  const int tile_n = blockIdx.x * 64;
  const int tile_m = blockIdx.y * 128;
  const int t = threadIdx.x;
  const int wave = t >> 5, lane = t & 31;

  const _Float16* gA = A + (size_t)(tile_m + t) * DIM;          // one row / thread
  const int rB = t >> 1, oB = (t & 1) * 16;
  const _Float16* gW = W + (size_t)(tile_n + rB) * DIM + oB;    // half row / thread
  const unsigned laBase[2] = { lds_off(&sA[0][t * 32]),       lds_off(&sA[1][t * 32]) };
  const unsigned lbBase[2] = { lds_off(&sB[0][rB * 32 + oB]), lds_off(&sB[1][rB * 32 + oB]) };

  v8f acc[2][4];
#pragma unroll
  for (int mt = 0; mt < 2; ++mt)
#pragma unroll
    for (int ct = 0; ct < 4; ++ct) acc[mt][ct] = vzero8();

  // prologue: stage k-tile 0 into buffer 0 (shared bases + immediate offsets)
#pragma unroll
  for (int i = 0; i < 4; ++i) ASYNC_B128(laBase[0], (const void*)gA, i * 16);
  ASYNC_B128(lbBase[0], (const void*)gW, 0);
  ASYNC_B128(lbBase[0], (const void*)gW, 16);

  for (int ki = 0; ki < DIM / 32; ++ki) {
    const int buf = ki & 1;
    if (ki + 1 < DIM / 32) {
      const _Float16* gak = gA + (ki + 1) * 32;
      const _Float16* gwk = gW + (ki + 1) * 32;
#pragma unroll
      for (int i = 0; i < 4; ++i) ASYNC_B128(laBase[buf ^ 1], (const void*)gak, i * 16);
      ASYNC_B128(lbBase[buf ^ 1], (const void*)gwk, 0);
      ASYNC_B128(lbBase[buf ^ 1], (const void*)gwk, 16);
      WAIT_ASYNC(6);   // in-order completion => current buf's 6 copies landed
    } else {
      WAIT_ASYNC(0);
    }
    __syncthreads();   // all waves' copies for `buf` visible
#pragma unroll
    for (int mt = 0; mt < 2; ++mt) {
      v16h fa = frag_a(sA[buf], wave * 32 + mt * 16, 32);
#pragma unroll
      for (int ct = 0; ct < 4; ++ct) {
        v16h fb = frag_b(sB[buf], ct * 16, 32);
        acc[mt][ct] = wmma16(fa, fb, acc[mt][ct]);
      }
    }
    __syncthreads();   // all waves done reading `buf` before it is refilled
  }

  // C layout: VGPR r -> row r (+8 for lanes 16..31), col = lane&15.
  const int cn = lane & 15;
  const int rh = (lane >> 4) * 8;
#pragma unroll
  for (int mt = 0; mt < 2; ++mt)
#pragma unroll
    for (int ct = 0; ct < 4; ++ct) {
      int col = tile_n + ct * 16 + cn;
      float bv = bias[col];
#pragma unroll
      for (int r8 = 0; r8 < 8; ++r8) {
        int row = tile_m + wave * 32 + mt * 16 + rh + r8;
        float v = acc[mt][ct][r8] + bv;
        if (MODE == 0) {
          int b = row >> 11, n = row & (SEQ - 1);
          int h = col >> 6,  d = col & 63;
          outh[(((size_t)(b * HEADS + h)) * SEQ + n) * DKH + d] = (_Float16)v;
        } else {
          outf[(size_t)row * DIM + col] = v;
        }
      }
    }
}

// ---------------- Flash-style cross attention --------------------------------
// grid = (SEQ/64, B*H); block = 128 (4 waves, 16 query rows each).
// Q,K,V: f16 [B,H,N,DK]; Out: f16 merged-heads [B,N,DIM].
// Q tile and K tiles staged via async global->LDS; V goes through VGPRs (it
// needs a transpose into LDS for the PV WMMA B-operand).
__global__ __launch_bounds__(128) void attn_kernel(
    const _Float16* __restrict__ Q, const _Float16* __restrict__ K,
    const _Float16* __restrict__ V, _Float16* __restrict__ Out)
{
  __shared__ alignas(32) _Float16 sQ[64 * 64];
  __shared__ alignas(32) _Float16 sK[64 * 64];
  __shared__ alignas(32) _Float16 sVt[64 * 64];  // transposed: [d][j]
  __shared__ alignas(32) _Float16 sP[64 * 64];
  __shared__ float sS[64 * 64];
  __shared__ float sAlpha[64];
  __shared__ float sL[64];

  const int bh = blockIdx.y;
  const int q0 = blockIdx.x * 64;
  const int t = threadIdx.x, wave = t >> 5, lane = t & 31;
  const _Float16* Qb = Q + (size_t)bh * SEQ * DKH;
  const _Float16* Kb = K + (size_t)bh * SEQ * DKH;
  const _Float16* Vb = V + (size_t)bh * SEQ * DKH;

  const int rT = t >> 1, oT = (t & 1) * 32;            // tile-load coords
  const unsigned lQ = lds_off(&sQ[rT * 64 + oT]);
  const unsigned lK = lds_off(&sK[rT * 64 + oT]);

  {  // async-stage Q tile once: 64 rows x 64 halfs
    const _Float16* src = Qb + (size_t)(q0 + rT) * DKH + oT;
#pragma unroll
    for (int i = 0; i < 4; ++i) ASYNC_B128(lQ, (const void*)src, i * 16);
  }

  v8f oacc[4];
#pragma unroll
  for (int i = 0; i < 4; ++i) oacc[i] = vzero8();
  float m_row = -__builtin_inff(), l_row = 0.0f;
  const int cn = lane & 15, rh = (lane >> 4) * 8;

  for (int kt = 0; kt < SEQ / 64; ++kt) {
    const int kv0 = kt * 64;
    {  // K tile async; V tile through VGPRs, transposed into sVt
      const _Float16* ksrc = Kb + (size_t)(kv0 + rT) * DKH + oT;
#pragma unroll
      for (int i = 0; i < 4; ++i) ASYNC_B128(lK, (const void*)ksrc, i * 16);
      const uint4* vsrc = (const uint4*)(Vb + (size_t)(kv0 + rT) * DKH + oT);
      _Float16 tmp[32];
#pragma unroll
      for (int i = 0; i < 4; ++i) ((uint4*)tmp)[i] = vsrc[i];
#pragma unroll
      for (int i = 0; i < 32; ++i) sVt[(oT + i) * 64 + rT] = tmp[i];
    }
    WAIT_ASYNC(0);      // Q (first iter) + this K tile are in LDS
    __syncthreads();

    // S = Q @ K^T   (per wave: 16 queries x 64 keys, k over DK=64)
    v8f sacc[4];
#pragma unroll
    for (int i = 0; i < 4; ++i) sacc[i] = vzero8();
#pragma unroll
    for (int kc = 0; kc < 2; ++kc) {
      v16h fq = frag_a(sQ + kc * 32, wave * 16, 64);
#pragma unroll
      for (int jt = 0; jt < 4; ++jt) {
        v16h fk = frag_b(sK + kc * 32, jt * 16, 64);
        sacc[jt] = wmma16(fq, fk, sacc[jt]);
      }
    }
#pragma unroll
    for (int jt = 0; jt < 4; ++jt)
#pragma unroll
      for (int r8 = 0; r8 < 8; ++r8)
        sS[(wave * 16 + rh + r8) * 64 + jt * 16 + cn] = sacc[jt][r8] * 0.125f;
    __syncthreads();

    {  // online softmax: 2 lanes per row, 32 cols each
      int row = wave * 16 + (lane & 15);
      int ch  = (lane >> 4) * 32;
      const float* srow = &sS[row * 64 + ch];
      float mx = -__builtin_inff();
#pragma unroll
      for (int i = 0; i < 32; ++i) mx = fmaxf(mx, srow[i]);
      mx = fmaxf(mx, __shfl_xor(mx, 16, 32));
      float m_new = fmaxf(m_row, mx);
      float sum = 0.0f;
#pragma unroll
      for (int i = 0; i < 32; ++i) {
        float p = __expf(srow[i] - m_new);
        sum += p;
        sP[row * 64 + ch + i] = (_Float16)p;
      }
      sum += __shfl_xor(sum, 16, 32);
      float alpha = __expf(m_row - m_new);
      l_row = l_row * alpha + sum;
      m_row = m_new;
      if (lane < 16) sAlpha[row] = alpha;
    }
    __syncthreads();

#pragma unroll
    for (int r8 = 0; r8 < 8; ++r8) {
      float a = sAlpha[wave * 16 + rh + r8];
#pragma unroll
      for (int dt = 0; dt < 4; ++dt) oacc[dt][r8] *= a;
    }
    // O += P @ V   (k over kv-index = 64)
#pragma unroll
    for (int jc = 0; jc < 2; ++jc) {
      v16h fp = frag_a(sP + jc * 32, wave * 16, 64);
#pragma unroll
      for (int dt = 0; dt < 4; ++dt) {
        v16h fv = frag_b(sVt + jc * 32, dt * 16, 64);
        oacc[dt] = wmma16(fp, fv, oacc[dt]);
      }
    }
    __syncthreads();
  }

  if (lane < 16) sL[wave * 16 + lane] = l_row;
  __syncthreads();

  const int b = bh / HEADS, h = bh % HEADS;
#pragma unroll
  for (int dt = 0; dt < 4; ++dt)
#pragma unroll
    for (int r8 = 0; r8 < 8; ++r8) {
      int rloc = wave * 16 + rh + r8;
      float val = oacc[dt][r8] * (1.0f / sL[rloc]);
      int n = q0 + rloc;
      Out[((size_t)b * SEQ + n) * DIM + h * DKH + dt * 16 + cn] = (_Float16)val;
    }
}

// ---------------- Orchestration ----------------------------------------------
extern "C" void kernel_launch(void* const* d_in, const int* in_sizes, int n_in,
                              void* d_out, int out_size, void* d_ws, size_t ws_size,
                              hipStream_t stream) {
  (void)in_sizes; (void)n_in; (void)out_size; (void)ws_size;
  const float* text   = (const float*)d_in[0];
  const float* vision = (const float*)d_in[1];
  const float* ln1w = (const float*)d_in[2];
  const float* ln1b = (const float*)d_in[3];
  const float* ln2w = (const float*)d_in[4];
  const float* ln2b = (const float*)d_in[5];
  const float* Wf32[8]; const float* bf32[8];
  for (int i = 0; i < 8; ++i) {        // order: q1,k1,v1,q2,k2,v2,o1,o2
    Wf32[i] = (const float*)d_in[6 + 2 * i];
    bf32[i] = (const float*)d_in[7 + 2 * i];
  }

  const size_t T   = (size_t)MROWS * DIM;   // 3,145,728 halfs per activation
  const size_t WSZ = (size_t)DIM * DIM;     // 589,824 halfs per weight
  _Float16* tn = (_Float16*)d_ws;
  _Float16* vn = tn + T;
  _Float16* qkv[6];
  for (int i = 0; i < 6; ++i) qkv[i] = vn + T + (size_t)i * T;
  _Float16* a1 = qkv[5] + T;
  _Float16* a2 = a1 + T;
  _Float16* wh[8];
  for (int i = 0; i < 8; ++i) wh[i] = a2 + T + (size_t)i * WSZ;
  // total ws: 10*T + 8*WSZ halfs ~= 70 MB

  ln_f16_kernel<<<MROWS, 256, 0, stream>>>(text,   ln1w, ln1b, tn);
  ln_f16_kernel<<<MROWS, 256, 0, stream>>>(vision, ln2w, ln2b, vn);

  {
    int n4 = (int)(WSZ / 4);
    int blocks = (n4 + 255) / 256;
    for (int i = 0; i < 8; ++i)
      cvt_f16x4_kernel<<<blocks, 256, 0, stream>>>((const float4*)Wf32[i], wh[i], n4);
  }

  dim3 gg(DIM / 64, MROWS / 128);
  for (int i = 0; i < 3; ++i)  // tq,tk,tv
    gemm_wmma_kernel<0><<<gg, 128, 0, stream>>>(tn, wh[i], bf32[i], qkv[i], nullptr);
  for (int i = 3; i < 6; ++i)  // vq,vk,vv
    gemm_wmma_kernel<0><<<gg, 128, 0, stream>>>(vn, wh[i], bf32[i], qkv[i], nullptr);

  dim3 ag(SEQ / 64, NBATCH * HEADS);
  attn_kernel<<<ag, 128, 0, stream>>>(qkv[0], qkv[4], qkv[5], a1);  // tq · (vk,vv)
  attn_kernel<<<ag, 128, 0, stream>>>(qkv[3], qkv[1], qkv[2], a2);  // vq · (tk,tv)

  float* out = (float*)d_out;
  gemm_wmma_kernel<1><<<gg, 128, 0, stream>>>(a1, wh[6], bf32[6], nullptr, out);
  gemm_wmma_kernel<1><<<gg, 128, 0, stream>>>(a2, wh[7], bf32[7], nullptr, out + T);
}
